// ins_U_cluloss_64622077935621
// MI455X (gfx1250) — compile-verified, matrix operand order checked by reading, not run on hardware
//
#include <hip/hip_runtime.h>
#include <stdint.h>

// ---------------- problem constants ----------------
#define C_CLUST  256
#define B_ROWS   16384
#define K_NN     4096
#define N_FEAT   128
#define INV_TEMP 2.0f          // 1/TEMP, TEMP=0.5 (folded into z_i A-fragments)

// ---------------- tiling ----------------
#define NTHREADS 256           // 8 waves (wave32)
#define NWAVE    8
#define WROWS    16            // rows of z_i per wave (one WMMA M-tile)
#define TB       (NWAVE*WROWS) // 128 rows per block
#define CH_ROWS  64            // streamed B-operand rows per chunk (4 col tiles)
#define STR      136           // LDS row stride in bf16 (128 + 8 pad -> no bank conflict)

typedef __bf16 bf16_t;
typedef bf16_t v16bf __attribute__((ext_vector_type(16)));
typedef float  v8f   __attribute__((ext_vector_type(8)));

// f32 -> bf16, native conversion (lowers to hardware cvt on gfx1250)
static __device__ inline bf16_t f2bf(float f) { return (bf16_t)f; }

// Build a 16x32 bf16 A-fragment (optionally pre-scaled) from a row-major f32 matrix.
// ISA 16-bit A layout: lane L holds row M=L%16; elements 0..7 cover
// K = kb + 8*(L/16) + {0..7}, elements 8..15 cover K = kb + 16 + 8*(L/16) + {0..7}.
static __device__ inline v16bf load_afrag(const float* __restrict__ rowptr, int kb, int half,
                                          float scale) {
  const float4* q0 = (const float4*)(rowptr + kb + 8 * half);
  const float4* q1 = (const float4*)(rowptr + kb + 16 + 8 * half);
  float4 x0 = q0[0], x1 = q0[1], y0 = q1[0], y1 = q1[1];
  v16bf v;
  v[0]=f2bf(x0.x*scale);  v[1]=f2bf(x0.y*scale);  v[2]=f2bf(x0.z*scale);  v[3]=f2bf(x0.w*scale);
  v[4]=f2bf(x1.x*scale);  v[5]=f2bf(x1.y*scale);  v[6]=f2bf(x1.z*scale);  v[7]=f2bf(x1.w*scale);
  v[8]=f2bf(y0.x*scale);  v[9]=f2bf(y0.y*scale);  v[10]=f2bf(y0.z*scale); v[11]=f2bf(y0.w*scale);
  v[12]=f2bf(y1.x*scale); v[13]=f2bf(y1.y*scale); v[14]=f2bf(y1.z*scale); v[15]=f2bf(y1.w*scale);
  return v;
}

// Two 16x16 output tiles at once: preload all B fragments (16 ds_load_b128 -> one
// wait), then 8 WMMAs as two independent accumulator chains for XDL back-to-back issue.
static __device__ inline void mm_tile_pair(const v16bf a[4], const bf16_t* __restrict__ buf,
                                           int tp, int lane, v8f& acc0, v8f& acc1) {
  const int half = lane >> 4;
  const int n0   = (tp * 2) * 16 + (lane & 15);   // output col = streamed row
  const bf16_t* b0 = buf + n0 * STR + 8 * half;
  const bf16_t* b1 = b0 + 16 * STR;
  union U { uint4 u[2]; v16bf v; };
  U B0[4], B1[4];
  #pragma unroll
  for (int kk = 0; kk < 4; ++kk) {
    const uint4* q0 = (const uint4*)(b0 + kk * 32); // q[0]: elems 0..7, q[2]: +16 K offset
    const uint4* q1 = (const uint4*)(b1 + kk * 32);
    B0[kk].u[0] = q0[0]; B0[kk].u[1] = q0[2];
    B1[kk].u[0] = q1[0]; B1[kk].u[1] = q1[2];
  }
  v8f z = {};
  acc0 = z; acc1 = z;
  #pragma unroll
  for (int kk = 0; kk < 4; ++kk) {
    acc0 = __builtin_amdgcn_wmma_f32_16x16x32_bf16(false, a[kk], false, B0[kk].v,
                                                   (short)0, acc0, false, false);
    acc1 = __builtin_amdgcn_wmma_f32_16x16x32_bf16(false, a[kk], false, B1[kk].v,
                                                   (short)0, acc1, false, false);
  }
}

// ---- producer: 64x128 f32 chunk -> bf16 LDS tile (double buffered) ----
static __device__ inline void load_chunk_regs(float4 r[8], const float* __restrict__ src, int ch) {
  const float4* s4 = (const float4*)(src + (size_t)ch * CH_ROWS * N_FEAT);
  const int tid = threadIdx.x;
  #pragma unroll
  for (int q = 0; q < 8; ++q) r[q] = s4[q * NTHREADS + tid];
}

static __device__ inline void store_chunk_lds(bf16_t* __restrict__ buf, const float4 r[8]) {
  const int tid = threadIdx.x;
  #pragma unroll
  for (int q = 0; q < 8; ++q) {
    int f   = q * NTHREADS + tid;   // float4 index within chunk
    int row = f >> 5;               // 32 float4 per 128-wide row
    int col = (f & 31) << 2;
    union { uint2 u2; bf16_t b[4]; } pk;
    pk.b[0] = f2bf(r[q].x); pk.b[1] = f2bf(r[q].y);
    pk.b[2] = f2bf(r[q].z); pk.b[3] = f2bf(r[q].w);
    *(uint2*)(buf + row * STR + col) = pk.u2;     // 8B aligned: 272|8, col*2|8
  }
}

// ---- per-element updates (acc elem j <-> row M = j + 8*(lane/16), col N = lane%16) ----
static __device__ inline void argmax_update(const v8f& acc, int ctg, int lane,
                                            float bv[8], int bi[8]) {
  const int col = ctg * 16 + (lane & 15);
  #pragma unroll
  for (int j = 0; j < 8; ++j) {
    float x = acc[j];
    if (x > bv[j]) { bv[j] = x; bi[j] = col; }    // strict '>' keeps first index
  }
}

// x already includes the 1/TEMP factor (folded into A fragments)
template<bool CENT>
static __device__ inline void sm_update(const v8f& acc, int ctg, int lane,
                                        float m[8], float s[8], float p[8],
                                        const int mcj[8]) {
  const int col = ctg * 16 + (lane & 15);
  #pragma unroll
  for (int j = 0; j < 8; ++j) {
    float x  = acc[j];
    float mo = m[j];
    float mn = fmaxf(mo, x);
    float sc = __expf(mo - mn);
    float w  = __expf(x - mn);
    // D==16 == tile width: positive nn block of row b is exactly col-tile mc[b]
    bool pos = CENT ? (col == mcj[j]) : (ctg == mcj[j]);
    s[j] = s[j] * sc + w;
    p[j] = p[j] * sc + (pos ? w : 0.0f);
    m[j] = mn;
  }
}

// ---- streamed GEMM phase: double-buffered LDS pipeline over `nch` 64-row chunks ----
// Pipeline: iter ch stores chunk ch+1 (buffer freed by last barrier) FIRST, so the
// ds_stores drain under the WMMAs; global loads run two chunks ahead.
// MODE 0: argmax (z_orl vs z_cent); MODE 1: cent softmax; MODE 2: nn softmax
template<int MODE>
static __device__ inline void stream_phase(const float* __restrict__ src, int nch,
                                           bf16_t (*buf)[CH_ROWS * STR],
                                           const v16bf a[4], int lane,
                                           float* bv, int* bi,
                                           float* m, float* s, float* p, const int* mcj) {
  float4 r[8];
  load_chunk_regs(r, src, 0);
  store_chunk_lds(buf[0], r);
  if (nch > 1) load_chunk_regs(r, src, 1);        // chunk 1 in flight across barrier
  __syncthreads();
  for (int ch = 0; ch < nch; ++ch) {
    if (ch + 1 < nch) store_chunk_lds(buf[(ch + 1) & 1], r);  // overlaps WMMAs below
    if (ch + 2 < nch) load_chunk_regs(r, src, ch + 2);
    if (ch + 3 < nch)
      __builtin_prefetch(src + (size_t)(ch + 3) * CH_ROWS * N_FEAT + threadIdx.x * 32, 0, 0);
    #pragma unroll
    for (int tp = 0; tp < 2; ++tp) {              // 2 tile pairs per chunk
      v8f acc0, acc1;
      mm_tile_pair(a, buf[ch & 1], tp, lane, acc0, acc1);
      const int ctg = ch * 4 + tp * 2;
      if constexpr (MODE == 0) {
        argmax_update(acc0, ctg,     lane, bv, bi);
        argmax_update(acc1, ctg + 1, lane, bv, bi);
      } else if constexpr (MODE == 1) {
        sm_update<true >(acc0, ctg,     lane, m, s, p, mcj);
        sm_update<true >(acc1, ctg + 1, lane, m, s, p, mcj);
      } else {
        sm_update<false>(acc0, ctg,     lane, m, s, p, mcj);
        sm_update<false>(acc1, ctg + 1, lane, m, s, p, mcj);
      }
    }
    __syncthreads();
  }
}

__global__ __launch_bounds__(NTHREADS)
void cluloss_main(const float* __restrict__ z_cent, const float* __restrict__ z_orl,
                  const float* __restrict__ z_nn,   const float* __restrict__ z_i,
                  float* __restrict__ d_out,        // [0] loss (reducer), [1..B] max_center
                  float* __restrict__ partials) {
  __shared__ bf16_t buf[2][CH_ROWS * STR];          // ~34.8 KB double buffer
  __shared__ int    mc_sh[TB];
  __shared__ float  wsum[2 * NWAVE];

  const int tid  = threadIdx.x;
  const int lane = tid & 31;
  const int w    = tid >> 5;
  const int half = lane >> 4;
  const int rb   = blockIdx.x * TB;
  const int myrow = rb + w * WROWS + (lane & 15);

  // A fragments live in registers for the whole kernel (z_i, pre-scaled by 1/TEMP)
  // and for phase 1 (z_orl, unscaled: argmax is scale-invariant).
  v16bf a_i[4], a_o[4];
  {
    const float* zi_row = z_i   + (size_t)myrow * N_FEAT;
    const float* zo_row = z_orl + (size_t)myrow * N_FEAT;
    #pragma unroll
    for (int kk = 0; kk < 4; ++kk) {
      a_i[kk] = load_afrag(zi_row, kk * 32, half, INV_TEMP);
      a_o[kk] = load_afrag(zo_row, kk * 32, half, 1.0f);
    }
  }

  // ---------- Phase 1: hard cluster assignment (argmax over C) ----------
  float bv[8]; int bi[8];
  #pragma unroll
  for (int j = 0; j < 8; ++j) { bv[j] = -1e30f; bi[j] = 0; }
  stream_phase<0>(z_cent, C_CLUST / CH_ROWS, buf, a_o, lane,
                  bv, bi, nullptr, nullptr, nullptr, nullptr);

  // cross-lane argmax over the 16 column-lanes of each half (wave32 shuffles)
  #pragma unroll
  for (int off = 1; off < 16; off <<= 1) {
    #pragma unroll
    for (int j = 0; j < 8; ++j) {
      float ov = __shfl_xor(bv[j], off, 32);
      int   oi = __shfl_xor(bi[j], off, 32);
      if (ov > bv[j] || (ov == bv[j] && oi < bi[j])) { bv[j] = ov; bi[j] = oi; }
    }
  }
  if ((lane & 15) == 0) {
    #pragma unroll
    for (int j = 0; j < 8; ++j) {
      int r = w * WROWS + j + 8 * half;
      mc_sh[r] = bi[j];
      d_out[1 + rb + r] = (float)bi[j];             // argmax output (exact in f32)
    }
  }
  __syncthreads();

  int mcj[8];
  #pragma unroll
  for (int j = 0; j < 8; ++j) mcj[j] = mc_sh[w * WROWS + j + 8 * half];

  // ---------- Phase 2: fused exp-sums with online max rescaling ----------
  float m[8], s[8], p[8];
  #pragma unroll
  for (int j = 0; j < 8; ++j) { m[j] = -1e30f; s[j] = 0.0f; p[j] = 0.0f; }

  stream_phase<1>(z_cent, C_CLUST / CH_ROWS, buf, a_i, lane,
                  nullptr, nullptr, m, s, p, mcj);
  stream_phase<2>(z_nn,   K_NN    / CH_ROWS, buf, a_i, lane,
                  nullptr, nullptr, m, s, p, mcj);

  // merge (m,s,p) across the 16 column-lanes with max rescale
  #pragma unroll
  for (int off = 1; off < 16; off <<= 1) {
    #pragma unroll
    for (int j = 0; j < 8; ++j) {
      float om = __shfl_xor(m[j], off, 32);
      float os = __shfl_xor(s[j], off, 32);
      float op = __shfl_xor(p[j], off, 32);
      float mn  = fmaxf(m[j], om);
      float sc1 = __expf(m[j] - mn);
      float sc2 = __expf(om - mn);
      s[j] = s[j] * sc1 + os * sc2;
      p[j] = p[j] * sc1 + op * sc2;
      m[j] = mn;
    }
  }

  // -log(pos/neg) = log(s) - log(p); exp(m) cancels. Deterministic block reduce.
  if ((lane & 15) == 0) {
    float acc = 0.0f;
    #pragma unroll
    for (int j = 0; j < 8; ++j) acc += __logf(s[j]) - __logf(p[j]);
    wsum[w * 2 + half] = acc;
  }
  __syncthreads();
  if (tid == 0) {
    float t = 0.0f;
    #pragma unroll
    for (int i = 0; i < 2 * NWAVE; ++i) t += wsum[i];
    partials[blockIdx.x] = t;
  }
}

__global__ __launch_bounds__(256)
void cluloss_reduce(const float* __restrict__ partials, float* __restrict__ out,
                    int nparts, float invB) {
  __shared__ float sh[256];
  const int t = threadIdx.x;
  sh[t] = (t < nparts) ? partials[t] : 0.0f;
  __syncthreads();
  for (int s2 = 128; s2 > 0; s2 >>= 1) {
    if (t < s2) sh[t] += sh[t + s2];
    __syncthreads();
  }
  if (t == 0) out[0] = sh[0] * invB;
}

extern "C" void kernel_launch(void* const* d_in, const int* in_sizes, int n_in,
                              void* d_out, int out_size, void* d_ws, size_t ws_size,
                              hipStream_t stream) {
  const float* z_cent = (const float*)d_in[0];
  const float* z_orl  = (const float*)d_in[1];
  const float* z_nn   = (const float*)d_in[2];
  const float* z_i    = (const float*)d_in[3];
  float* out      = (float*)d_out;
  float* partials = (float*)d_ws;                  // 128 floats of scratch

  const int nblocks = B_ROWS / TB;                 // 128
  cluloss_main<<<dim3(nblocks), dim3(NTHREADS), 0, stream>>>(
      z_cent, z_orl, z_nn, z_i, out, partials);
  cluloss_reduce<<<dim3(1), dim3(256), 0, stream>>>(
      partials, out, nblocks, 1.0f / (float)B_ROWS);
}
